// LSTMBlock_46153718563146
// MI455X (gfx1250) — matrix operand order repaired and checked by reading, not compiled, not run on hardware
//
#include <hip/hip_runtime.h>

typedef __bf16 bf16_t;
typedef __attribute__((ext_vector_type(16))) __bf16 v16bf;
typedef __attribute__((ext_vector_type(8)))  float  v8f;

// ---------------------------------------------------------------------------
// Workspace layout (bytes). All intermediates are stored in WMMA-fragment
// order so compute kernels use contiguous 32B per-lane vector accesses.
//   xc  f32 frag-order [128 img][16 y][16 nt][32 ln][8 r]   : 33,554,432
//   WhB bf16 packed    [9 tap][2 kb][16 nt][32 ln][16 e]    :    294,912
//   WxB bf16 packed    [16 nt][32 ln][16 e]                 :     16,384
//   hA  bf16 perm-chan [4096 pos][4 grp][16 s]              :    524,288
//   hB  bf16 perm-chan [4096 pos][4 grp][16 s]              :    524,288
//   c   f32 frag-order [16 b][16 y][4 ht][32 ln][8 r]       :  1,048,576
// ---------------------------------------------------------------------------
#define OFF_XC   0
#define OFF_WHB  33554432u
#define OFF_WXB  (OFF_WHB + 294912u)
#define OFF_HA   (OFF_WXB + 16384u)
#define OFF_HB   (OFF_HA + 524288u)
#define OFF_C    (OFF_HB + 524288u)

// h channel permutation: for channel ch (0..63), the A-fragment for
// (kb = ch>>5, half) reads 16 contiguous bf16 at group = kb*2+half:
//   slot s<8  -> ch = kb*32 + 8*half + s
//   slot s>=8 -> ch = kb*32 + 16 + 8*half + (s-8)
__device__ __forceinline__ int h_perm(int ch) {
    int kb = ch >> 5, w = ch & 31;
    int half = (w >> 3) & 1;
    int s = (w & 16) ? ((w & 7) + 8) : (w & 7);
    return (kb * 2 + half) * 16 + s;
}

// Pack Wh (3,3,64,256) f32 and Wx (3,3,3,256) f32 into bf16 WMMA B-fragments.
// B-fragment per-lane layout (16x16x32 bf16): N = lane%16, K(e) = e + (e>=8?8:0) + 8*(lane/16).
__global__ void convlstm_pack_weights(const float* __restrict__ Wh,
                                      const float* __restrict__ Wx,
                                      bf16_t* __restrict__ WhB,
                                      bf16_t* __restrict__ WxB) {
    int tid = blockIdx.x * blockDim.x + threadIdx.x;
    const int NWH = 9 * 2 * 16 * 32;
    if (tid < NWH) {
        int lane = tid & 31;
        int nt   = (tid >> 5) & 15;
        int kb   = (tid >> 9) & 1;
        int tap  = tid >> 10;                      // 0..8
        int n    = nt * 16 + (lane & 15);
        bf16_t* dst = WhB + ((size_t)tid) * 16;
        #pragma unroll
        for (int e = 0; e < 16; ++e) {
            int kin = kb * 32 + e + ((e >= 8) ? 8 : 0) + 8 * (lane >> 4); // input channel 0..63
            dst[e] = (bf16_t)Wh[(tap * 64 + kin) * 256 + n];
        }
    } else if (tid < NWH + 16 * 32) {
        int t2   = tid - NWH;
        int lane = t2 & 31;
        int nt   = t2 >> 5;
        int n    = nt * 16 + (lane & 15);
        bf16_t* dst = WxB + ((size_t)t2) * 16;
        #pragma unroll
        for (int e = 0; e < 16; ++e) {
            int k = e + ((e >= 8) ? 8 : 0) + 8 * (lane >> 4); // im2col k = (dy*3+dx)*3+c, 27 real
            dst[e] = (k < 27) ? (bf16_t)Wx[k * 256 + n] : (bf16_t)0.0f;
        }
    }
}

// Stride-16 3x3x3 input conv + bias: x(128,256,256,3) -> xc in fragment order.
// One wave per (img, ho) row tile: M = wo (16), loops over 16 N-tiles, K=32.
// 2048 waves = 256 blocks x 8 waves.
__global__ void convlstm_input_conv(const float* __restrict__ x,
                                    const bf16_t* __restrict__ WxB,
                                    const float* __restrict__ bias,
                                    float* __restrict__ xc) {
    int wave = blockIdx.x * 8 + (threadIdx.x >> 5);   // 0..2047
    int lane = threadIdx.x & 31;
    int img  = wave >> 4;                             // 0..127 (b*8+t)
    int ho   = wave & 15;
    int wo   = lane & 15;
    int half = lane >> 4;

    // A fragment: M = wo, K(e) = e + (e>=8?8:0) + 8*half (ISA 16-bit A layout).
    v16bf a = {};
    #pragma unroll
    for (int e = 0; e < 16; ++e) {
        int K = e + ((e >= 8) ? 8 : 0) + 8 * half;
        float v = 0.0f;
        if (K < 27) {
            int dy = K / 9, rr = K % 9;
            int dx = rr / 3, c = rr % 3;
            // SAME pad for stride16/k3 on 256 is zero padding -> always in range.
            v = x[(((size_t)img * 256 + ho * 16 + dy) * 256 + wo * 16 + dx) * 3 + c];
        }
        a[e] = (bf16_t)v;
    }

    #pragma unroll 1
    for (int nt = 0; nt < 16; ++nt) {
        v16bf bm = *(const v16bf*)(WxB + ((size_t)(nt * 32 + lane)) * 16);
        v8f acc = {};
        acc = __builtin_amdgcn_wmma_f32_16x16x32_bf16(false, a, false, bm,
                                                      (short)0, acc, false, false);
        float bv = bias[nt * 16 + (lane & 15)];
        #pragma unroll
        for (int r = 0; r < 8; ++r) acc[r] += bv;
        // Fragment-order store: one contiguous 32B vector per lane.
        *(v8f*)(xc + (((size_t)img * 16 + ho) * 16 + nt) * 256 + lane * 8) = acc;
    }
}

__device__ __forceinline__ float hsig(float z) {
    return fminf(fmaxf(0.2f * z + 0.5f, 0.0f), 1.0f);
}

// One ConvLSTM step. All 8 waves of a block share the same hidden-channel tile
// ht, so the block needs exactly 72 B-fragments (72KB). These are staged into
// LDS once via global_load_async_to_lds_b128 (ASYNCcnt), then the fully
// unrolled 9-tap x 2-kb x 4-gate chain issues 72 v_wmma_f32_16x16x32_bf16 with
// B operands from low-latency ds_load_b128. Gate math is wave-local.
// 128 blocks x 256 threads = 1024 waves.
__global__ void convlstm_step(const float* __restrict__ xc,
                              const bf16_t* __restrict__ WhB,
                              const bf16_t* __restrict__ hprev,
                              bf16_t* __restrict__ hnext,
                              float* __restrict__ cstate,
                              int t,
                              float* __restrict__ out) {
    __shared__ __align__(64) bf16_t lds_wh[18 * 4 * 32 * 16];  // 72KB

    int tid  = threadIdx.x;
    int lane = tid & 31;
    int wv   = tid >> 5;
    int ht   = blockIdx.x & 3;                        // hidden tile, same per block
    int grp  = blockIdx.x >> 2;                       // 0..31
    int tile = grp * 8 + wv;                          // 0..255
    int b    = tile >> 4;
    int y    = tile & 15;
    int m    = lane & 15;                             // output x position
    int half = lane >> 4;
    int img  = b * 8 + t;

    // ---- Async-stage this block's 72 Wh B-fragments into LDS --------------
    // 72 frags x 1KB = 4608 16B chunks; 18 chunks per thread.
    {
        unsigned long long src = (unsigned long long)(uintptr_t)WhB;
        #pragma unroll
        for (int j = 0; j < 18; ++j) {
            int c      = j * 256 + tid;               // chunk id 0..4607
            int f      = c >> 6;                      // local fragment 0..71
            int within = c & 63;                      // 16B chunk in fragment
            int t2k    = f >> 2;                      // tap*2+kb
            int g      = f & 3;                       // gate
            unsigned src_off = (unsigned)(((t2k * 16 + g * 4 + ht) * 512 + within * 8) * 2);
            unsigned lds_off = (unsigned)(uintptr_t)&lds_wh[f * 512 + within * 8];
            asm volatile("global_load_async_to_lds_b128 %0, %1, %2"
                         :: "v"(lds_off), "v"(src_off), "s"(src)
                         : "memory");
        }
        asm volatile("s_wait_asynccnt 0" ::: "memory");
    }
    __syncthreads();

    // Init accumulators from fragment-order xc (bias folded in): 32B per gate.
    v8f acc[4];
    #pragma unroll
    for (int g = 0; g < 4; ++g) {
        acc[g] = *(const v8f*)(xc +
            (((size_t)img * 16 + y) * 16 + (g * 4 + ht)) * 256 + lane * 8);
    }

    #pragma unroll
    for (int dy = 0; dy < 3; ++dy) {
        int yy = y + dy - 1;
        bool yok = (yy >= 0) && (yy < 16);
        #pragma unroll
        for (int dx = 0; dx < 3; ++dx) {
            int tap = dy * 3 + dx;
            int xx  = m + dx - 1;                  // per-lane neighbor column
            bool inb = yok && (xx >= 0) && (xx < 16);
            const bf16_t* hp = hprev + ((size_t)(b * 256 + yy * 16 + xx)) * 64;
            #pragma unroll
            for (int kb = 0; kb < 2; ++kb) {
                // Permuted-channel h: whole A fragment = one contiguous 32B load,
                // lane-zeroed at SAME-pad borders (reconverges before WMMA).
                v16bf a = {};
                if (inb) a = *(const v16bf*)(hp + (kb * 2 + half) * 16);
                #pragma unroll
                for (int g = 0; g < 4; ++g) {
                    int fl = (tap * 2 + kb) * 4 + g;
                    v16bf bm = *(const v16bf*)&lds_wh[(fl * 32 + lane) * 16];
                    acc[g] = __builtin_amdgcn_wmma_f32_16x16x32_bf16(
                        false, a, false, bm, (short)0, acc[g], false, false);
                }
            }
        }
    }

    // Gate math (Keras defaults: hard-sigmoid recurrent act., linear cell act.).
    // c in fragment order -> 32B vector read-modify-write per lane.
    float* cp = cstate + (((size_t)(b * 16 + y) * 4 + ht) * 32 + lane) * 8;
    v8f cprev = *(const v8f*)cp;
    v8f cnew, hval;
    #pragma unroll
    for (int r = 0; r < 8; ++r) {
        float i = hsig(acc[0][r]);
        float f = hsig(acc[1][r]);
        float o = hsig(acc[3][r]);
        float cn = f * cprev[r] + i * acc[2][r];   // acc[2] = g (linear)
        cnew[r] = cn;
        hval[r] = o * cn;
    }
    *(v8f*)cp = cnew;

    // h store: transpose D-order (lane = channel) -> permuted-channel h layout.
    int ch = ht * 16 + m;
    int pc = h_perm(ch);
    #pragma unroll
    for (int r = 0; r < 8; ++r) {
        int xpos = r + 8 * half;                   // D: M = r + 8*(lane/16)
        size_t pos = ((size_t)b * 16 + y) * 16 + xpos;
        hnext[pos * 64 + pc] = (bf16_t)hval[r];
        if (out) out[pos * 64 + ch] = hval[r];     // reference layout, t=7 only
    }
}

extern "C" void kernel_launch(void* const* d_in, const int* in_sizes, int n_in,
                              void* d_out, int out_size, void* d_ws, size_t ws_size,
                              hipStream_t stream) {
    const float* x    = (const float*)d_in[0];  // (16,8,256,256,3)
    const float* Wx   = (const float*)d_in[1];  // (3,3,3,256)
    const float* Wh   = (const float*)d_in[2];  // (3,3,64,256)
    const float* bias = (const float*)d_in[3];  // (256,)

    char* ws = (char*)d_ws;
    float*  xc  = (float*) (ws + OFF_XC);
    bf16_t* WhB = (bf16_t*)(ws + OFF_WHB);
    bf16_t* WxB = (bf16_t*)(ws + OFF_WXB);
    bf16_t* hA  = (bf16_t*)(ws + OFF_HA);
    bf16_t* hB  = (bf16_t*)(ws + OFF_HB);
    float*  cst = (float*) (ws + OFF_C);

    // h0 = 0, c0 = 0 every call (zero is layout-invariant; capturable memsets).
    hipMemsetAsync(hA, 0, 524288u, stream);
    hipMemsetAsync(cst, 0, 1048576u, stream);

    convlstm_pack_weights<<<(9 * 2 * 16 * 32 + 16 * 32 + 255) / 256, 256, 0, stream>>>(
        Wh, Wx, WhB, WxB);

    convlstm_input_conv<<<256, 256, 0, stream>>>(x, WxB, bias, xc);

    bf16_t* hcur = hA;
    bf16_t* hnxt = hB;
    for (int t = 0; t < 8; ++t) {
        convlstm_step<<<128, 256, 0, stream>>>(
            xc, WhB, hcur, hnxt, cst, t, (t == 7) ? (float*)d_out : nullptr);
        bf16_t* tmp = hcur; hcur = hnxt; hnxt = tmp;
    }
}
